// TransformerAttention_3487513445005
// MI455X (gfx1250) — matrix operand order repaired and checked by reading, not compiled, not run on hardware
//
#include <hip/hip_runtime.h>
#include <hip/hip_bf16.h>
#include <math.h>

// ---------------- problem constants ----------------
#define B_   2
#define T_   2048
#define D_   2048
#define H_   16
#define KV_  4
#define DH_  128
#define WIN_ 1024
#define NMETA_ 16

typedef __bf16 bf16_t;
typedef __attribute__((ext_vector_type(16))) __bf16 v16bf;
typedef __attribute__((ext_vector_type(8)))  __bf16 v8bf;
typedef __attribute__((ext_vector_type(8)))  float  v8f;
typedef __attribute__((ext_vector_type(4)))  unsigned uint4v;
typedef __attribute__((ext_vector_type(8)))  unsigned uint8v;

__device__ __forceinline__ bf16_t f2bf(float f) {
  unsigned u = __builtin_bit_cast(unsigned, f);
  u += 0x7fffu + ((u >> 16) & 1u);          // round-to-nearest-even
  unsigned short h = (unsigned short)(u >> 16);
  return __builtin_bit_cast(bf16_t, h);
}
__device__ __forceinline__ unsigned short f2bfbits(float f) {
  return __builtin_bit_cast(unsigned short, f2bf(f));
}

// LDS aperture: low 32 bits of a generic pointer to __shared__ are the
// wave-relative LDS byte address (ISA 10.2 aperture rules).
__device__ __forceinline__ unsigned lds_addr32(const void* p) {
  return (unsigned)(uintptr_t)p;
}

// GLOBAL_LOAD_ASYNC_TO_LDS_B128: per-lane 16B global->LDS, tracked by ASYNCcnt.
__device__ __forceinline__ void async_ld_b128(unsigned ldsaddr, const void* g) {
  unsigned long long ga = (unsigned long long)(uintptr_t)g;
  asm volatile("global_load_async_to_lds_b128 %0, %1, off"
               :: "v"(ldsaddr), "v"(ga) : "memory");
}
__device__ __forceinline__ void wait_async_le8() {
  asm volatile("s_wait_asynccnt 8" ::: "memory");
}
__device__ __forceinline__ void wait_async_le0() {
  asm volatile("s_wait_asynccnt 0" ::: "memory");
}

// ---- TDM: one 2D tile DMA per block k-slice (TENSOR_LOAD_TO_LDS, §7/§8) ----
// group0: [1:0]=count=1, [63:32]=lds_addr, [120:64]=global_addr, [127:126]=type=2
__device__ __forceinline__ uint4v tdm_g0(unsigned ldsaddr, const void* g) {
  unsigned long long ga = (unsigned long long)(uintptr_t)g;
  uint4v g0;
  g0[0] = 1u;
  g0[1] = ldsaddr;
  g0[2] = (unsigned)ga;
  g0[3] = ((unsigned)(ga >> 32) & 0x01FFFFFFu) | 0x80000000u;
  return g0;
}
__device__ __forceinline__ void tdm_load(uint4v g0, uint8v g1) {
  asm volatile("tensor_load_to_lds %0, %1" :: "s"(g0), "s"(g1) : "memory");
}

// ---------------- casts ----------------
__global__ __launch_bounds__(256)
void cast_f32_bf16(const float* __restrict__ in, bf16_t* __restrict__ out, long n4) {
  long i = (long)blockIdx.x * blockDim.x + threadIdx.x;
  if (i >= n4) return;
  float4 v = ((const float4*)in)[i];
  unsigned long long pk =  (unsigned long long)f2bfbits(v.x)
                        | ((unsigned long long)f2bfbits(v.y) << 16)
                        | ((unsigned long long)f2bfbits(v.z) << 32)
                        | ((unsigned long long)f2bfbits(v.w) << 48);
  ((unsigned long long*)out)[i] = pk;
}

// LDS-tiled transpose-cast: in [R][C] f32 -> out [C][R] bf16, 64x64 tiles.
__global__ __launch_bounds__(256)
void transpose_cast(const float* __restrict__ in, bf16_t* __restrict__ out, int R, int C) {
  __shared__ bf16_t tile[64][65];             // +1 pad: dodge bank conflicts
  const int tx = threadIdx.x & 63;
  const int ty = threadIdx.x >> 6;            // 0..3
  const int bc = blockIdx.x * 64;             // C base
  const int br = blockIdx.y * 64;             // R base
  #pragma unroll
  for (int rr = ty; rr < 64; rr += 4)
    tile[rr][tx] = f2bf(in[(size_t)(br + rr) * C + bc + tx]);
  __syncthreads();
  #pragma unroll
  for (int rr = ty; rr < 64; rr += 4)
    out[(size_t)(bc + rr) * R + br + tx] = tile[tx][rr];
}

// ---------------- WMMA GEMM: 64x64 per wave, 64x256 per block ----------------
// A slice (64x32) staged by TDM (one tensor_load_to_lds per slice, wave 0).
// B slice (256 cols x 32 k) staged per-thread by async-to-LDS, double buffered.
// MODE 0: Q -> rope + 1/sqrt(Dh), store [B][H][T][Dh]
// MODE 1: K -> rope,             store [B][KV][T][Dh]
// MODE 2: V ->                   store [B][KV][Dh][T]
// MODE 3: out projection -> f32  store [M][N]
template<int MODE>
__global__ __launch_bounds__(128)
void gemm_wmma(const bf16_t* __restrict__ A, const bf16_t* __restrict__ Bt,
               void* __restrict__ Out, int M, int K, int N)
{
  __shared__ alignas(16) unsigned char Atile[2][64 * 64];     // 2 x 4KB
  __shared__ alignas(16) unsigned char Btile[2][256 * 64];    // 2 x 16KB

  const int tid  = threadIdx.x;
  const int lane = tid & 31;
  const int w    = tid >> 5;
  const int hf   = lane >> 4;
  const int l16  = lane & 15;

  const int nblk = N >> 8;                    // N / 256
  const int bm = blockIdx.x / nblk;
  const int bn = blockIdx.x - bm * nblk;
  const int m0 = bm * 64;
  const int n0 = bn * 256 + w * 64;

  // TDM descriptor group1 (static): 2B elems, tensor K x M, tile 32 x 64, stride K
  uint8v g1;
  g1[0] = 0x00010000u;                                         // data_size = 1 (2B)
  g1[1] = ((unsigned)K & 0xFFFFu) << 16;                       // tensor_dim0 lo
  g1[2] = (((unsigned)K >> 16) & 0xFFFFu) | (((unsigned)M & 0xFFFFu) << 16);
  g1[3] = (((unsigned)M >> 16) & 0xFFFFu) | (32u << 16);       // tile_dim0 = 32
  g1[4] = 64u;                                                 // tile_dim1 = 64
  g1[5] = (unsigned)K;                                         // dim0 stride lo
  g1[6] = 0u;
  g1[7] = 0u;
  const unsigned ldsA[2] = { lds_addr32(&Atile[0][0]), lds_addr32(&Atile[1][0]) };
  const bf16_t* gA = A + (size_t)m0 * K;                       // A tile base

  // B async staging: this thread owns columns (tid*2, tid*2+1), 4 B128 each.
  const bf16_t* gB0 = Bt + (size_t)(bn * 256 + tid * 2) * K;
  const bf16_t* gB1 = gB0 + K;
  unsigned ldsB0[2], ldsB1[2];
  #pragma unroll
  for (int p = 0; p < 2; ++p) {
    ldsB0[p] = lds_addr32(&Btile[p][(tid * 2) * 64]);
    ldsB1[p] = ldsB0[p] + 64;
  }

  v8f acc[4][4];
  #pragma unroll
  for (int ti = 0; ti < 4; ++ti)
    #pragma unroll
    for (int tj = 0; tj < 4; ++tj)
      acc[ti][tj] = (v8f){0.f,0.f,0.f,0.f,0.f,0.f,0.f,0.f};

  const int nk = K >> 5;
  // prologue: stage slice 0 into buffer 0
  #pragma unroll
  for (int c = 0; c < 4; ++c) {
    async_ld_b128(ldsB0[0] + c * 16, gB0 + c * 8);
    async_ld_b128(ldsB1[0] + c * 16, gB1 + c * 8);
  }
  if (w == 0) tdm_load(tdm_g0(ldsA[0], gA), g1);

  for (int kk = 0; kk < nk; ++kk) {
    const int p = kk & 1;
    if (kk + 1 < nk) {                        // stage next slice
      #pragma unroll
      for (int c = 0; c < 4; ++c) {
        async_ld_b128(ldsB0[1 - p] + c * 16, gB0 + (kk + 1) * 32 + c * 8);
        async_ld_b128(ldsB1[1 - p] + c * 16, gB1 + (kk + 1) * 32 + c * 8);
      }
      if (w == 0) tdm_load(tdm_g0(ldsA[1 - p], gA + (kk + 1) * 32), g1);
      wait_async_le8();                       // current B slice landed
      if (w == 0) __builtin_amdgcn_s_wait_tensorcnt(1);   // current A slice landed
    } else {
      wait_async_le0();
      if (w == 0) __builtin_amdgcn_s_wait_tensorcnt(0);
    }
    __syncthreads();

    v16bf af[4];
    #pragma unroll
    for (int ti = 0; ti < 4; ++ti) {
      const unsigned char* rb = &Atile[p][(ti * 16 + l16) * 64];
      ((v8bf*)&af[ti])[0] = *(const v8bf*)(rb + hf * 32);
      ((v8bf*)&af[ti])[1] = *(const v8bf*)(rb + hf * 32 + 16);
    }
    v16bf bfr[4];
    #pragma unroll
    for (int tj = 0; tj < 4; ++tj)
      bfr[tj] = *(const v16bf*)(&Btile[p][(w * 64 + tj * 16 + l16) * 64 + hf * 32]);

    #pragma unroll
    for (int ti = 0; ti < 4; ++ti)
      #pragma unroll
      for (int tj = 0; tj < 4; ++tj)
        acc[ti][tj] = __builtin_amdgcn_wmma_f32_16x16x32_bf16(
            false, af[ti], false, bfr[tj], (short)0, acc[ti][tj], false, false);

    __syncthreads();
  }

  // ---------------- epilogues ----------------
  if (MODE == 0 || MODE == 1) {
    const int heads = (MODE == 0) ? H_ : KV_;
    bf16_t* outp = (bf16_t*)Out;
    #pragma unroll
    for (int tj = 0; tj < 4; ++tj) {
      const int n  = n0 + tj * 16 + l16;
      const int hd = n >> 7;
      const int d  = n & (DH_ - 1);
      const int de = d & ~1;
      const float inv = __expf(-(float)de * (9.210340371976184f / (float)DH_));
      #pragma unroll
      for (int ti = 0; ti < 4; ++ti) {
        #pragma unroll
        for (int j = 0; j < 8; ++j) {
          const int m = m0 + ti * 16 + j + hf * 8;
          const int b = m >> 11;
          const int t = m & (T_ - 1);
          float s, c;
          __sincosf((float)t * inv, &s, &c);
          float own = acc[ti][tj][j];
          float par = __shfl_xor(own, 1, 32);       // rope partner lane (d^1)
          float o = (d & 1) ? (par * s + own * c) : (own * c - par * s);
          if (MODE == 0) o *= 0.08838834764831845f; // 1/sqrt(Dh) folded into Q
          outp[((size_t)(b * heads + hd) * T_ + t) * DH_ + d] = f2bf(o);
        }
      }
    }
  } else if (MODE == 2) {
    bf16_t* outp = (bf16_t*)Out;
    #pragma unroll
    for (int tj = 0; tj < 4; ++tj) {
      const int n  = n0 + tj * 16 + l16;
      const int hd = n >> 7;
      const int d  = n & (DH_ - 1);
      #pragma unroll
      for (int ti = 0; ti < 4; ++ti) {
        #pragma unroll
        for (int j = 0; j < 8; ++j) {
          const int m = m0 + ti * 16 + j + hf * 8;
          const int b = m >> 11;
          const int t = m & (T_ - 1);
          outp[((size_t)(b * KV_ + hd) * DH_ + d) * T_ + t] = f2bf(acc[ti][tj][j]);
        }
      }
    }
  } else {
    float* outp = (float*)Out;
    #pragma unroll
    for (int tj = 0; tj < 4; ++tj) {
      const int n = n0 + tj * 16 + l16;
      #pragma unroll
      for (int ti = 0; ti < 4; ++ti) {
        #pragma unroll
        for (int j = 0; j < 8; ++j) {
          const int m = m0 + ti * 16 + j + hf * 8;
          outp[(size_t)m * N + n] = acc[ti][tj][j];
        }
      }
    }
  }
}

// ---------------- flash attention (sliding window + sinks) ----------------
__global__ __launch_bounds__(128)
void flash_swa(const bf16_t* __restrict__ qb, const bf16_t* __restrict__ kb,
               const bf16_t* __restrict__ vt, bf16_t* __restrict__ ob)
{
  __shared__ unsigned short Pb[4][16][32];     // per-wave P tile (bf16 bits)
  const int lane = threadIdx.x & 31;
  const int hf   = lane >> 4;
  const int l16  = lane & 15;
  const int w    = threadIdx.x >> 5;

  const int wid = blockIdx.x * 4 + w;
  const int qtph = T_ / 16;
  const int b   = wid / (H_ * qtph);
  const int rem = wid - b * (H_ * qtph);
  const int h   = rem / qtph;
  const int qt  = rem - h * qtph;
  const int q0  = qt * 16;
  const int kvh = h >> 2;                      // REP = 4

  const bf16_t* Q  = qb + (size_t)(b * H_  + h)   * T_  * DH_;
  const bf16_t* Kp = kb + (size_t)(b * KV_ + kvh) * T_  * DH_;
  const bf16_t* Vt = vt + (size_t)(b * KV_ + kvh) * DH_ * T_;   // [Dh][T]

  v16bf qf[4];
  {
    const bf16_t* qp = Q + (size_t)(q0 + l16) * DH_;
    #pragma unroll
    for (int c = 0; c < 4; ++c) {
      ((v8bf*)&qf[c])[0] = *(const v8bf*)(qp + c * 32 + hf * 16);
      ((v8bf*)&qf[c])[1] = *(const v8bf*)(qp + c * 32 + hf * 16 + 8);
    }
  }

  const v8f vzero = {0.f,0.f,0.f,0.f,0.f,0.f,0.f,0.f};
  v8f o[8];
  float mi[8], li[8];
  #pragma unroll
  for (int j = 0; j < 8; ++j) { o[j] = vzero; mi[j] = -INFINITY; li[j] = 0.f; }

  int start = 32;
  const int lo = q0 - (WIN_ - 1);
  if (lo > 32) start = lo & ~31;

  for (int k0 = 0; k0 < q0 + 16; k0 = (k0 == 0 ? start : k0 + 32)) {
    const int kn = (k0 == 0 ? start : k0 + 32);
    if (kn < q0 + 16) {
      __builtin_prefetch((const void*)(Kp + (size_t)(kn + l16) * DH_), 0, 3);
      __builtin_prefetch((const void*)(Kp + (size_t)(kn + 16 + l16) * DH_), 0, 3);
      __builtin_prefetch((const void*)(Vt + (size_t)l16 * T_ + kn), 0, 3);
    }

    // ---- S = Q K^T for keys [k0, k0+32) ----
    v8f s0 = vzero, s1 = vzero;
    #pragma unroll
    for (int c = 0; c < 4; ++c) {
      v16bf kf0 = *(const v16bf*)(Kp + (size_t)(k0 + l16)      * DH_ + c * 32 + hf * 16);
      v16bf kf1 = *(const v16bf*)(Kp + (size_t)(k0 + 16 + l16) * DH_ + c * 32 + hf * 16);
      s0 = __builtin_amdgcn_wmma_f32_16x16x32_bf16(false, qf[c], false, kf0, (short)0, s0, false, false);
      s1 = __builtin_amdgcn_wmma_f32_16x16x32_bf16(false, qf[c], false, kf1, (short)0, s1, false, false);
    }

    // ---- mask + online softmax ----
    const int ka  = k0 + l16;
    const int kb2 = k0 + 16 + l16;
    float alpha[8];
    #pragma unroll
    for (int j = 0; j < 8; ++j) {
      const int q = q0 + j + hf * 8;
      float va = s0[j], vb = s1[j];
      const bool aok = (ka  < NMETA_) || ((ka  <= q) && (q - ka  < WIN_));
      const bool bok = (kb2 < NMETA_) || ((kb2 <= q) && (q - kb2 < WIN_));
      va = aok ? va : -1e30f;
      vb = bok ? vb : -1e30f;
      float rm = fmaxf(va, vb);
      #pragma unroll
      for (int msk = 1; msk < 16; msk <<= 1) rm = fmaxf(rm, __shfl_xor(rm, msk, 32));
      const float mnew = fmaxf(mi[j], rm);
      const float al   = __expf(mi[j] - mnew);
      va = __expf(va - mnew);
      vb = __expf(vb - mnew);
      float rs = va + vb;
      #pragma unroll
      for (int msk = 1; msk < 16; msk <<= 1) rs += __shfl_xor(rs, msk, 32);
      li[j] = li[j] * al + rs;
      mi[j] = mnew;
      alpha[j] = al;
      const int row = j + hf * 8;
      Pb[w][row][l16]      = f2bfbits(va);
      Pb[w][row][l16 + 16] = f2bfbits(vb);
    }

    #pragma unroll
    for (int nt = 0; nt < 8; ++nt)
      #pragma unroll
      for (int j = 0; j < 8; ++j)
        o[nt][j] *= alpha[j];

    // ---- P @ V ----
    v16bf pf;
    ((v8bf*)&pf)[0] = *(const v8bf*)&Pb[w][l16][hf * 16];
    ((v8bf*)&pf)[1] = *(const v8bf*)&Pb[w][l16][hf * 16 + 8];
    #pragma unroll
    for (int nt = 0; nt < 8; ++nt) {
      v16bf vf = *(const v16bf*)(Vt + (size_t)(nt * 16 + l16) * T_ + k0 + hf * 16);
      o[nt] = __builtin_amdgcn_wmma_f32_16x16x32_bf16(false, pf, false, vf, (short)0, o[nt], false, false);
    }
  }

  // ---- finalize ----
  const size_t obase = ((size_t)(b * T_ + q0)) * (H_ * DH_) + (size_t)h * DH_;
  #pragma unroll
  for (int nt = 0; nt < 8; ++nt) {
    #pragma unroll
    for (int j = 0; j < 8; ++j) {
      const int row = j + hf * 8;
      const float val = o[nt][j] * (1.0f / li[j]);
      ob[obase + (size_t)row * (H_ * DH_) + nt * 16 + l16] = f2bf(val);
    }
  }
}

// ---------------- launch ----------------
extern "C" void kernel_launch(void* const* d_in, const int* in_sizes, int n_in,
                              void* d_out, int out_size, void* d_ws, size_t ws_size,
                              hipStream_t stream) {
  const float* x  = (const float*)d_in[0];
  const float* wq = (const float*)d_in[1];
  const float* wk = (const float*)d_in[2];
  const float* wv = (const float*)d_in[3];
  const float* wo = (const float*)d_in[4];

  char* ws = (char*)d_ws;
  const size_t OFF_XB  = 0;               // 8388608 bf16
  const size_t OFF_WQT = 16777216;        // 4194304 bf16
  const size_t OFF_WKT = 25165824;        // 1048576 bf16
  const size_t OFF_WVT = 27262976;        // 1048576 bf16
  const size_t OFF_WOT = 29360128;        // 4194304 bf16
  const size_t OFF_QB  = 37748736;        // 8388608 bf16
  const size_t OFF_KB  = 54525952;        // 2097152 bf16
  const size_t OFF_VT  = 58720256;        // 2097152 bf16
  const size_t OFF_OB  = 62914560;        // 8388608 bf16

  bf16_t* xb  = (bf16_t*)(ws + OFF_XB);
  bf16_t* wqT = (bf16_t*)(ws + OFF_WQT);
  bf16_t* wkT = (bf16_t*)(ws + OFF_WKT);
  bf16_t* wvT = (bf16_t*)(ws + OFF_WVT);
  bf16_t* woT = (bf16_t*)(ws + OFF_WOT);
  bf16_t* qb  = (bf16_t*)(ws + OFF_QB);
  bf16_t* kbf = (bf16_t*)(ws + OFF_KB);
  bf16_t* vtb = (bf16_t*)(ws + OFF_VT);
  bf16_t* obf = (bf16_t*)(ws + OFF_OB);

  const long nx4 = (long)B_ * T_ * D_ / 4;
  cast_f32_bf16<<<(int)((nx4 + 255) / 256), 256, 0, stream>>>(x, xb, nx4);
  transpose_cast<<<dim3((H_*DH_)/64, D_/64),  256, 0, stream>>>(wq, wqT, D_, H_ * DH_);
  transpose_cast<<<dim3((KV_*DH_)/64, D_/64), 256, 0, stream>>>(wk, wkT, D_, KV_ * DH_);
  transpose_cast<<<dim3((KV_*DH_)/64, D_/64), 256, 0, stream>>>(wv, wvT, D_, KV_ * DH_);
  transpose_cast<<<dim3(D_/64, (H_*DH_)/64),  256, 0, stream>>>(wo, woT, H_ * DH_, D_);

  const int M = B_ * T_;                               // 4096
  gemm_wmma<0><<<(M / 64) * (H_ * DH_ / 256),  128, 0, stream>>>(xb, wqT, (void*)qb,  M, D_, H_ * DH_);
  gemm_wmma<1><<<(M / 64) * (KV_ * DH_ / 256), 128, 0, stream>>>(xb, wkT, (void*)kbf, M, D_, KV_ * DH_);
  gemm_wmma<2><<<(M / 64) * (KV_ * DH_ / 256), 128, 0, stream>>>(xb, wvT, (void*)vtb, M, D_, KV_ * DH_);

  flash_swa<<<(B_ * H_ * (T_ / 16)) / 4, 128, 0, stream>>>(qb, kbf, vtb, obf);

  gemm_wmma<3><<<(M / 64) * (D_ / 256), 128, 0, stream>>>(obf, woT, d_out, M, H_ * DH_, D_);
}
